// CausalSelfAttention_6064493822470
// MI455X (gfx1250) — matrix operand order repaired and checked by reading, not compiled, not run on hardware
//
#include <hip/hip_runtime.h>
#include <hip/hip_bf16.h>

// ---------------- problem constants ----------------
#define BB   4
#define TT   2048
#define CC   512
#define HH   8
#define DD   64
#define NOBJ 4
#define NEG_BIG (-1.0e30f)

typedef __attribute__((ext_vector_type(16))) __bf16 bf16x16;
typedef __attribute__((ext_vector_type(8)))  __bf16 bf16x8;
typedef __attribute__((ext_vector_type(8)))  float  f32x8;

// ---------------- WMMA helpers ----------------
__device__ __forceinline__ f32x8 wmma_bf16(bf16x16 a, bf16x16 b, f32x8 c) {
  // D = A(16x32) * B(32x16) + C  -> v_wmma_f32_16x16x32_bf16
  return __builtin_amdgcn_wmma_f32_16x16x32_bf16(false, a, false, b, (short)0, c,
                                                 false, false);
}

// A fragment (16x32, row-major source, ld in elements).
// lane<16: m=lane, k = {0..7, 16..23};  lane>=16: m=lane-16, k = {8..15, 24..31}
__device__ __forceinline__ bf16x16 load_a_frag(const __bf16* src, int ld, int lane) {
  const int m  = lane & 15;
  const int kh = (lane >> 4) << 3;          // 0 or 8
  const __bf16* p = src + m * ld + kh;
  bf16x8 lo = *(const bf16x8*)(p);          // k = kh .. kh+7
  bf16x8 hi = *(const bf16x8*)(p + 16);     // k = kh+16 .. kh+23
  bf16x16 r;
#pragma unroll
  for (int i = 0; i < 8; ++i) { r[i] = lo[i]; r[i + 8] = hi[i]; }
  return r;
}

// B fragment (32x16).  Memory holds src[n][k] (i.e. B transposed, row = n).
// lane holds column n = lane&15; lane<16: k = {0..7, 16..23}; lane>=16: k = {8..15, 24..31}.
__device__ __forceinline__ bf16x16 load_b_frag(const __bf16* src, int ld, int lane) {
  const int n  = lane & 15;
  const int kh = (lane >> 4) << 3;          // 0 or 8
  const __bf16* p = src + n * ld + kh;
  bf16x8 lo = *(const bf16x8*)(p);
  bf16x8 hi = *(const bf16x8*)(p + 16);
  bf16x16 r;
#pragma unroll
  for (int i = 0; i < 8; ++i) { r[i] = lo[i]; r[i + 8] = hi[i]; }
  return r;
}

__device__ __forceinline__ void wave_lds_fence() {
  asm volatile("s_wait_dscnt 0" ::: "memory");
}

// ---------------- kernel 1: fp32 -> bf16 convert ----------------
__global__ void cvt_f32_bf16(const float* __restrict__ src, __bf16* __restrict__ dst, int n) {
  int i = blockIdx.x * blockDim.x + threadIdx.x;
  if (i < n) dst[i] = (__bf16)src[i];
}

// ---------------- kernel 2: fused QKV projection ----------------
// Block tile: 128 rows x 64 cols; 8 waves, each owning a 32x32 sub-tile
// (2 A-frags x 2 B-frags -> 4 WMMAs / k-step for better FLOPs-per-byte).
// blockIdx.x: N tile (C/64), blockIdx.y: M tile (B*T/128), blockIdx.z: 0=Q,1=K,2=V
__global__ __launch_bounds__(256) void qkv_gemm_kernel(
    const __bf16* __restrict__ xb,
    const __bf16* __restrict__ Wqb, const __bf16* __restrict__ Wkb,
    const __bf16* __restrict__ Wvb,
    const float* __restrict__ bq, const float* __restrict__ bk,
    const float* __restrict__ bv,
    __bf16* __restrict__ Q, __bf16* __restrict__ K, __bf16* __restrict__ Vt) {
  const int lane = threadIdx.x & 31;
  const int wave = threadIdx.x >> 5;
  const int mat  = blockIdx.z;

  const __bf16* W    = (mat == 0) ? Wqb : (mat == 1) ? Wkb : Wvb;
  const float*  bias = (mat == 0) ? bq  : (mat == 1) ? bk  : bv;

  const int rowBase = blockIdx.y * 128 + (wave & 3) * 32;  // token row in [0, B*T)
  const int colBase = blockIdx.x * 64 + (wave >> 2) * 32;  // channel in [0, C)

  f32x8 acc00 = {}, acc01 = {}, acc10 = {}, acc11 = {};
  for (int kk = 0; kk < CC; kk += 32) {
    bf16x16 a0 = load_a_frag(xb + (size_t)(rowBase +  0) * CC + kk, CC, lane);
    bf16x16 a1 = load_a_frag(xb + (size_t)(rowBase + 16) * CC + kk, CC, lane);
    bf16x16 b0 = load_b_frag(W + (size_t)(colBase +  0) * CC + kk, CC, lane);
    bf16x16 b1 = load_b_frag(W + (size_t)(colBase + 16) * CC + kk, CC, lane);
    acc00 = wmma_bf16(a0, b0, acc00);
    acc01 = wmma_bf16(a0, b1, acc01);
    acc10 = wmma_bf16(a1, b0, acc10);
    acc11 = wmma_bf16(a1, b1, acc11);
  }

  const float scale = (mat == 0) ? 0.125f : 1.0f;  // fold 1/sqrt(D) into Q
  f32x8 accs[2][2] = {{acc00, acc01}, {acc10, acc11}};
#pragma unroll
  for (int rs = 0; rs < 2; ++rs) {
#pragma unroll
    for (int cs = 0; cs < 2; ++cs) {
      f32x8 acc = accs[rs][cs];
      const int col = colBase + cs * 16 + (lane & 15);
      const int h = col >> 6, d = col & 63;
      const float bcol = bias[col];
#pragma unroll
      for (int i = 0; i < 8; ++i) {
        const int row = rowBase + rs * 16 + ((lane >> 4) << 3) + i;
        const int b = row >> 11;       // / T
        const int t = row & (TT - 1);  // % T
        const float v = (acc[i] + bcol) * scale;
        if (mat == 2) {
          Vt[(((size_t)(b * HH + h) * DD) + d) * TT + t] = (__bf16)v;   // [B,H,D,T]
        } else {
          __bf16* dst = (mat == 0) ? Q : K;
          dst[(((size_t)(b * HH + h) * TT) + t) * DD + d] = (__bf16)v;  // [B,H,T,D]
        }
      }
    }
  }
}

// ---------------- kernel 3: flash attention ----------------
// blockIdx.x: query tile of 128 rows; blockIdx.y: (b*H + h). One wave = 16 rows.
__global__ __launch_bounds__(256) void flash_attn_kernel(
    const __bf16* __restrict__ Q, const __bf16* __restrict__ K,
    const __bf16* __restrict__ Vt, const int* __restrict__ mask,
    __bf16* __restrict__ y) {
  __shared__ float  Sbuf[8][16][64];
  __shared__ __bf16 Pbuf[8][16][64];
  __shared__ float  stat_m[8][16];
  __shared__ float  stat_l[8][16];
  __shared__ float  stat_a[8][16];
  __shared__ float  pmax[8][16][2];
  __shared__ float  psum[8][16][2];

  const int lane = threadIdx.x & 31;
  const int wave = threadIdx.x >> 5;
  const int bh = blockIdx.y;
  const int b = bh >> 3, h = bh & 7;
  const int qbase = blockIdx.x * 128 + wave * 16;

  const __bf16* Qh = Q  + (size_t)bh * TT * DD;
  const __bf16* Kh = K  + (size_t)bh * TT * DD;
  const __bf16* Vh = Vt + (size_t)bh * DD * TT;
  const int*    mk = mask + b * TT;

  const bf16x16 qa0 = load_a_frag(Qh + qbase * DD +  0, DD, lane);
  const bf16x16 qa1 = load_a_frag(Qh + qbase * DD + 32, DD, lane);

  f32x8 oacc[4] = {};
  if (lane < 16) { stat_m[wave][lane] = NEG_BIG; stat_l[wave][lane] = 0.f; }
  wave_lds_fence();

  const int mbase = (lane >> 4) << 3;          // row offset for C/D layout
  const int klimit = qbase + 16;               // frames are 4 wide; tiles are 16-aligned

  for (int kb = 0; kb < klimit; kb += 64) {
    // prefetch next key tile (global_prefetch_b8) while this one computes
    if (kb + 64 < klimit) {
      __builtin_prefetch(Kh + (size_t)(kb + 64 + (lane & 63)) * DD, 0, 1);
    }

    // ---- S = Q K^T  (16 x 64), contraction over D ----
#pragma unroll
    for (int n = 0; n < 4; ++n) {
      const int kn = kb + n * 16;
      f32x8 s = {};
      bf16x16 bk0 = load_b_frag(Kh + kn * DD +  0, DD, lane);
      bf16x16 bk1 = load_b_frag(Kh + kn * DD + 32, DD, lane);
      s = wmma_bf16(qa0, bk0, s);
      s = wmma_bf16(qa1, bk1, s);
#pragma unroll
      for (int i = 0; i < 8; ++i)
        Sbuf[wave][mbase + i][n * 16 + (lane & 15)] = s[i];
    }
    wave_lds_fence();

    // ---- masked online softmax: lane -> (row r, key half hh) ----
    const int r  = lane & 15;
    const int hh = lane >> 4;
    const int qf = (qbase + r) >> 2;           // query frame
    float lmax = NEG_BIG;
#pragma unroll
    for (int j = 0; j < 32; ++j) {
      const int kidx = kb + hh * 32 + j;
      const bool ok = ((kidx >> 2) <= qf) && (mk[kidx] != 0);
      float v = ok ? Sbuf[wave][r][hh * 32 + j] : NEG_BIG;
      Sbuf[wave][r][hh * 32 + j] = v;          // keep masked value for pass 2
      lmax = fmaxf(lmax, v);
    }
    pmax[wave][r][hh] = lmax;
    wave_lds_fence();
    const float tmax = fmaxf(pmax[wave][r][0], pmax[wave][r][1]);
    const float mold = stat_m[wave][r];
    const float mnew = fmaxf(mold, tmax);
    const float alpha = __expf(mold - mnew);
    float ls = 0.f;
#pragma unroll
    for (int j = 0; j < 32; ++j) {
      const float e = __expf(Sbuf[wave][r][hh * 32 + j] - mnew);  // 0 if masked
      ls += e;
      Pbuf[wave][r][hh * 32 + j] = (__bf16)e;
    }
    psum[wave][r][hh] = ls;
    wave_lds_fence();
    if (hh == 0) {
      stat_l[wave][r] = alpha * stat_l[wave][r] + psum[wave][r][0] + psum[wave][r][1];
      stat_m[wave][r] = mnew;
      stat_a[wave][r] = alpha;
    }
    wave_lds_fence();

    // ---- rescale running output ----
    float al[8];
#pragma unroll
    for (int i = 0; i < 8; ++i) al[i] = stat_a[wave][mbase + i];
#pragma unroll
    for (int n = 0; n < 4; ++n)
#pragma unroll
      for (int i = 0; i < 8; ++i) oacc[n][i] *= al[i];

    // ---- O += P V, contraction over 64 keys ----
    const bf16x16 pa0 = load_a_frag(&Pbuf[wave][0][0],  64, lane);
    const bf16x16 pa1 = load_a_frag(&Pbuf[wave][0][32], 64, lane);
#pragma unroll
    for (int n = 0; n < 4; ++n) {
      bf16x16 bv0 = load_b_frag(Vh + (size_t)(n * 16) * TT + kb,      TT, lane);
      bf16x16 bv1 = load_b_frag(Vh + (size_t)(n * 16) * TT + kb + 32, TT, lane);
      oacc[n] = wmma_bf16(pa0, bv0, oacc[n]);
      oacc[n] = wmma_bf16(pa1, bv1, oacc[n]);
    }
  }

  // ---- finalize: divide by row sums, store bf16 y[B,T,C] ----
  float inv[8];
#pragma unroll
  for (int i = 0; i < 8; ++i) inv[i] = 1.0f / stat_l[wave][mbase + i];
#pragma unroll
  for (int n = 0; n < 4; ++n) {
    const int col = h * 64 + n * 16 + (lane & 15);
#pragma unroll
    for (int i = 0; i < 8; ++i) {
      const int t = qbase + mbase + i;
      y[((size_t)(b * TT + t)) * CC + col] = (__bf16)(oacc[n][i] * inv[i]);
    }
  }
}

// ---------------- kernel 4: output projection (fp32 out) ----------------
// Same 32x32-per-wave tiling as the QKV GEMM.
__global__ __launch_bounds__(256) void out_proj_kernel(
    const __bf16* __restrict__ yb, const __bf16* __restrict__ Wpb,
    const float* __restrict__ bp, float* __restrict__ out) {
  const int lane = threadIdx.x & 31;
  const int wave = threadIdx.x >> 5;
  const int rowBase = blockIdx.y * 128 + (wave & 3) * 32;
  const int colBase = blockIdx.x * 64 + (wave >> 2) * 32;

  f32x8 acc00 = {}, acc01 = {}, acc10 = {}, acc11 = {};
  for (int kk = 0; kk < CC; kk += 32) {
    bf16x16 a0 = load_a_frag(yb + (size_t)(rowBase +  0) * CC + kk, CC, lane);
    bf16x16 a1 = load_a_frag(yb + (size_t)(rowBase + 16) * CC + kk, CC, lane);
    bf16x16 b0 = load_b_frag(Wpb + (size_t)(colBase +  0) * CC + kk, CC, lane);
    bf16x16 b1 = load_b_frag(Wpb + (size_t)(colBase + 16) * CC + kk, CC, lane);
    acc00 = wmma_bf16(a0, b0, acc00);
    acc01 = wmma_bf16(a0, b1, acc01);
    acc10 = wmma_bf16(a1, b0, acc10);
    acc11 = wmma_bf16(a1, b1, acc11);
  }
  f32x8 accs[2][2] = {{acc00, acc01}, {acc10, acc11}};
#pragma unroll
  for (int rs = 0; rs < 2; ++rs) {
#pragma unroll
    for (int cs = 0; cs < 2; ++cs) {
      f32x8 acc = accs[rs][cs];
      const int col = colBase + cs * 16 + (lane & 15);
      const float bcol = bp[col];
#pragma unroll
      for (int i = 0; i < 8; ++i) {
        const int row = rowBase + rs * 16 + ((lane >> 4) << 3) + i;
        out[(size_t)row * CC + col] = acc[i] + bcol;
      }
    }
  }
}

// ---------------- host launcher ----------------
extern "C" void kernel_launch(void* const* d_in, const int* in_sizes, int n_in,
                              void* d_out, int out_size, void* d_ws, size_t ws_size,
                              hipStream_t stream) {
  const float* x    = (const float*)d_in[0];
  const int*   mask = (const int*)  d_in[1];
  const float* Wq   = (const float*)d_in[2];
  const float* bq   = (const float*)d_in[3];
  const float* Wk   = (const float*)d_in[4];
  const float* bk   = (const float*)d_in[5];
  const float* Wv   = (const float*)d_in[6];
  const float* bv   = (const float*)d_in[7];
  const float* Wp   = (const float*)d_in[8];
  const float* bp   = (const float*)d_in[9];
  float* out = (float*)d_out;

  char* ws = (char*)d_ws;
  size_t off = 0;
  const size_t nX = (size_t)BB * TT * CC;   // 4,194,304
  const size_t nW = (size_t)CC * CC;        // 262,144
  __bf16* xb  = (__bf16*)(ws + off); off += nX * 2;
  __bf16* Wqb = (__bf16*)(ws + off); off += nW * 2;
  __bf16* Wkb = (__bf16*)(ws + off); off += nW * 2;
  __bf16* Wvb = (__bf16*)(ws + off); off += nW * 2;
  __bf16* Wpb = (__bf16*)(ws + off); off += nW * 2;
  __bf16* Qb  = (__bf16*)(ws + off); off += nX * 2;   // [B,H,T,D], pre-scaled
  __bf16* Kb  = (__bf16*)(ws + off); off += nX * 2;   // [B,H,T,D]
  __bf16* Vtb = (__bf16*)(ws + off); off += nX * 2;   // [B,H,D,T]
  __bf16* yb  = (__bf16*)(ws + off); off += nX * 2;   // [B,T,C]

  // 1) fp32 -> bf16
  cvt_f32_bf16<<<(int)((nX + 255) / 256), 256, 0, stream>>>(x, xb, (int)nX);
  cvt_f32_bf16<<<(int)((nW + 255) / 256), 256, 0, stream>>>(Wq, Wqb, (int)nW);
  cvt_f32_bf16<<<(int)((nW + 255) / 256), 256, 0, stream>>>(Wk, Wkb, (int)nW);
  cvt_f32_bf16<<<(int)((nW + 255) / 256), 256, 0, stream>>>(Wv, Wvb, (int)nW);
  cvt_f32_bf16<<<(int)((nW + 255) / 256), 256, 0, stream>>>(Wp, Wpb, (int)nW);

  // 2) QKV projections (z: 0=Q,1=K,2=V)
  qkv_gemm_kernel<<<dim3(CC / 64, (BB * TT) / 128, 3), 256, 0, stream>>>(
      xb, Wqb, Wkb, Wvb, bq, bk, bv, Qb, Kb, Vtb);

  // 3) flash attention
  flash_attn_kernel<<<dim3(TT / 128, BB * HH), 256, 0, stream>>>(
      Qb, Kb, Vtb, mask, yb);

  // 4) output projection
  out_proj_kernel<<<dim3(CC / 64, (BB * TT) / 128), 256, 0, stream>>>(
      yb, Wpb, bp, out);
}
